// DecoderBlock_54271206752308
// MI455X (gfx1250) — matrix operand order repaired and checked by reading, not compiled, not run on hardware
//
#include <hip/hip_runtime.h>
#include <hip/hip_bf16.h>
#include <cstdint>

// Decoder block for MI455X (gfx1250, wave32, WMMA).
// - All GEMMs on v_wmma_f32_16x16x32_bf16 (fp32 accumulation).
// - bf16 tile staging uses gfx1250 async DMA: global_load_async_to_lds_b128,
//   synchronized with s_wait_asynccnt (double-buffered, overlapped with WMMA).
// - fp32 weight tiles are prefetched into registers and converted to bf16.

#define DEVFN __device__ __forceinline__

typedef __attribute__((ext_vector_type(16))) __bf16 v16bf;
typedef __attribute__((ext_vector_type(8)))  float  v8f;

union Frag {
  v16bf v;
  unsigned int u32[8];
  unsigned short u16[16];
};

DEVFN unsigned short f2bf(float x) {
  union { float f; unsigned int u; } c; c.f = x;
  unsigned int r = c.u + 0x7FFFu + ((c.u >> 16) & 1u);  // round-to-nearest-even
  return (unsigned short)(r >> 16);
}
DEVFN float bf2f(unsigned short x) {
  union { float f; unsigned int u; } c; c.u = ((unsigned int)x) << 16;
  return c.f;
}

DEVFN v8f wmma_bf16(const Frag& a, const Frag& b, v8f c) {
  // (neg_a, A, neg_b, B, c_mod, C, reuse_a, reuse_b)
  return __builtin_amdgcn_wmma_f32_16x16x32_bf16(false, a.v, false, b.v,
                                                 (short)0, c, false, false);
}

// ---- gfx1250 async copy: 16 bytes global -> LDS, tracked by ASYNCcnt ----
// Flat pointers to __shared__ carry the LDS byte offset in the low 32 bits.
DEVFN unsigned lds_off(const void* p) { return (unsigned)(uintptr_t)p; }

DEVFN void async_ld16(unsigned lds, const void* g) {
  asm volatile("global_load_async_to_lds_b128 %0, %1, off"
               :: "v"(lds), "v"((unsigned long long)(uintptr_t)g)
               : "memory");
}
DEVFN void wait_async0() { asm volatile("s_wait_asynccnt 0x0" ::: "memory"); }

// Convert 16 f32 (4 x float4 in regs) -> 16 bf16, store 32B to LDS.
DEVFN void cvt_store16(const float4* fr, unsigned short* dst) {
  unsigned pk[8];
#pragma unroll
  for (int j = 0; j < 4; ++j) {
    pk[2 * j]     = (unsigned)f2bf(fr[j].x) | ((unsigned)f2bf(fr[j].y) << 16);
    pk[2 * j + 1] = (unsigned)f2bf(fr[j].z) | ((unsigned)f2bf(fr[j].w) << 16);
  }
  uint4* d = (uint4*)dst;
  d[0] = make_uint4(pk[0], pk[1], pk[2], pk[3]);
  d[1] = make_uint4(pk[4], pk[5], pk[6], pk[7]);
}

// A fragment (16x32 M x K); LDS stored row-major [m][k], stride in ushorts.
DEVFN void gather_a(const unsigned short* smem, int stride, int row0, int ks, Frag& f) {
  const int lane = threadIdx.x & 31;
  const int h = lane >> 4;
  const unsigned short* base = smem + (row0 + (lane & 15)) * stride + ks * 32;
#pragma unroll
  for (int v = 0; v < 8; ++v) {
    const int kb = 2 * v + 16 * h;
    f.u32[v] = *(const unsigned int*)(base + kb);
  }
}

// B fragment (32x16 K x N); LDS stored [n][k].
DEVFN void gather_b_nk(const unsigned short* smem, int stride, int n0, int ks, Frag& f) {
  const int lane = threadIdx.x & 31;
  const int h = lane >> 4;
  const unsigned short* base = smem + (n0 + (lane & 15)) * stride + ks * 32 + 16 * h;
#pragma unroll
  for (int v = 0; v < 8; ++v)
    f.u32[v] = *(const unsigned int*)(base + 2 * v);
}

// B fragment (32x16 K x N); LDS stored [k][n] (V tile stored [key][dk]).
DEVFN void gather_b_kn(const unsigned short* smem, int stride, int n0, int ks, Frag& f) {
  const int lane = threadIdx.x & 31;
  const int h = lane >> 4;
  const unsigned short* base = smem + (ks * 32 + 16 * h) * stride + n0 + (lane & 15);
#pragma unroll
  for (int v = 0; v < 8; ++v) {
    f.u16[2 * v]     = base[(2 * v) * stride];
    f.u16[2 * v + 1] = base[(2 * v + 1) * stride];
  }
}

// ---------------------------------------------------------------- RMSNorm
__global__ __launch_bounds__(256) void rmsnorm_kernel(const float* __restrict__ x,
                                                      const float* __restrict__ g,
                                                      unsigned short* __restrict__ out) {
  const int row = blockIdx.x;
  const int t = threadIdx.x;
  const float4 v = ((const float4*)(x + (size_t)row * 1024))[t];
  float ss = v.x * v.x + v.y * v.y + v.z * v.z + v.w * v.w;
#pragma unroll
  for (int off = 1; off < 32; off <<= 1) ss += __shfl_xor(ss, off, 32);
  __shared__ float red[8];
  if ((t & 31) == 0) red[t >> 5] = ss;
  __syncthreads();
  float tot = 0.f;
#pragma unroll
  for (int i = 0; i < 8; ++i) tot += red[i];
  const float scale = rsqrtf(tot * (1.0f / 1024.0f) + 1e-6f);
  const float4 gv = ((const float4*)g)[t];
  unsigned short* o = out + (size_t)row * 1024 + t * 4;
  o[0] = f2bf(v.x * scale * gv.x);
  o[1] = f2bf(v.y * scale * gv.y);
  o[2] = f2bf(v.z * scale * gv.z);
  o[3] = f2bf(v.w * scale * gv.w);
}

// ---------------------------------------------------------------- RoPE (in place, bf16)
__global__ __launch_bounds__(256) void rope_kernel(unsigned short* __restrict__ q,
                                                   unsigned short* __restrict__ k) {
  const int idx = blockIdx.x * 256 + threadIdx.x;   // (b,s,head,i) packed
  const int i = idx & 31;
  const int head = (idx >> 5) & 15;
  const int s = (idx >> 9) & 2047;
  const int b = idx >> 20;
  const size_t off = ((size_t)(b * 2048 + s)) * 1024 + head * 64 + i;
  const float freq = __powf(10000.0f, -(float)i * (1.0f / 32.0f));
  const float ang = (float)s * freq;
  const float c = cosf(ang), sn = sinf(ang);
  {
    const float x1 = bf2f(q[off]), x2 = bf2f(q[off + 32]);
    q[off]      = f2bf(x1 * c - x2 * sn);
    q[off + 32] = f2bf(x2 * c + x1 * sn);
  }
  {
    const float x1 = bf2f(k[off]), x2 = bf2f(k[off + 32]);
    k[off]      = f2bf(x1 * c - x2 * sn);
    k[off + 32] = f2bf(x2 * c + x1 * sn);
  }
}

// ---------------------------------------------------------------- Generic WMMA GEMM
// C[M,N] = A_bf16[M,K] @ W_f32[N,K]^T.   MODE 0: bf16 out.  MODE 1: f32 out + residual.
// Block tile 128x128, 8 waves (4x2), wave tile 32x64, K-step 32.
// Double-buffered: A via async-to-LDS DMA, B via register prefetch + convert.
template <int MODE>
__global__ __launch_bounds__(256) void gemm_bf16_kernel(
    const unsigned short* __restrict__ A, const float* __restrict__ Bw,
    void* __restrict__ Cout, const float* __restrict__ Res,
    int M, int N, int K) {
  __shared__ __align__(16) unsigned short As[2][128 * 40];
  __shared__ __align__(16) unsigned short Bs[2][128 * 40];
  const int t = threadIdx.x;
  const int lane = t & 31;
  const int wv = t >> 5;
  const int waveM = wv >> 1;   // 0..3
  const int waveN = wv & 1;    // 0..1
  const int h = lane >> 4, nlo = lane & 15;
  const int blockM = blockIdx.y * 128;
  const int blockN = blockIdx.x * 128;
  const int lrow = t >> 1;
  const int lcol = (t & 1) * 16;

  const unsigned short* gA = A + (size_t)(blockM + lrow) * K + lcol;
  const float*          gB = Bw + (size_t)(blockN + lrow) * K + lcol;
  const unsigned ldsA[2] = {lds_off(&As[0][lrow * 40 + lcol]),
                            lds_off(&As[1][lrow * 40 + lcol])};

  v8f acc[2][4];
#pragma unroll
  for (int mt = 0; mt < 2; ++mt)
#pragma unroll
    for (int nt = 0; nt < 4; ++nt)
#pragma unroll
      for (int r = 0; r < 8; ++r) acc[mt][nt][r] = 0.f;

  // prologue: stage tile 0
  async_ld16(ldsA[0], gA);
  async_ld16(ldsA[0] + 16, gA + 8);
  float4 fr[4];
  {
    const float4* s = (const float4*)gB;
    fr[0] = s[0]; fr[1] = s[1]; fr[2] = s[2]; fr[3] = s[3];
  }

  const int nk = K >> 5;
  for (int i = 0; i < nk; ++i) {
    const int cur = i & 1;
    cvt_store16(fr, &Bs[cur][lrow * 40 + lcol]);
    wait_async0();
    __syncthreads();
    if (i + 1 < nk) {  // issue next tile while computing this one
      const int nxt = (i + 1) << 5;
      async_ld16(ldsA[cur ^ 1], gA + nxt);
      async_ld16(ldsA[cur ^ 1] + 16, gA + nxt + 8);
      const float4* s = (const float4*)(gB + nxt);
      fr[0] = s[0]; fr[1] = s[1]; fr[2] = s[2]; fr[3] = s[3];
    }
    Frag a0, a1;
    gather_a(As[cur], 40, waveM * 32, 0, a0);
    gather_a(As[cur], 40, waveM * 32 + 16, 0, a1);
#pragma unroll
    for (int nt = 0; nt < 4; ++nt) {
      Frag bb;
      gather_b_nk(Bs[cur], 40, waveN * 64 + nt * 16, 0, bb);
      acc[0][nt] = wmma_bf16(a0, bb, acc[0][nt]);
      acc[1][nt] = wmma_bf16(a1, bb, acc[1][nt]);
    }
    __syncthreads();
  }
#pragma unroll
  for (int mt = 0; mt < 2; ++mt)
#pragma unroll
    for (int nt = 0; nt < 4; ++nt)
#pragma unroll
      for (int r = 0; r < 8; ++r) {
        const int row = blockM + waveM * 32 + mt * 16 + r + 8 * h;
        const int col = blockN + waveN * 64 + nt * 16 + nlo;
        const size_t idx = (size_t)row * N + col;
        const float v = acc[mt][nt][r];
        if (MODE == 0) ((unsigned short*)Cout)[idx] = f2bf(v);
        else           ((float*)Cout)[idx] = v + Res[idx];
      }
}

// ---------------------------------------------------------------- w1 GEMM fused with SwiGLU
// g[M,DFF] = u1 * silu(u2), u1 = A @ W1[0:DFF]^T, u2 = A @ W1[DFF:2DFF]^T.
__global__ __launch_bounds__(256) void gemm_swiglu_kernel(
    const unsigned short* __restrict__ A, const float* __restrict__ W1,
    unsigned short* __restrict__ G, int M, int K) {
  __shared__ __align__(16) unsigned short As[2][128 * 40];
  __shared__ __align__(16) unsigned short B1s[2][128 * 40];
  __shared__ __align__(16) unsigned short B2s[2][128 * 40];
  const int t = threadIdx.x;
  const int lane = t & 31;
  const int wv = t >> 5;
  const int waveM = wv >> 1;
  const int waveN = wv & 1;
  const int h = lane >> 4, nlo = lane & 15;
  const int blockM = blockIdx.y * 128;
  const int blockN = blockIdx.x * 128;
  const int lrow = t >> 1;
  const int lcol = (t & 1) * 16;

  const unsigned short* gA = A + (size_t)(blockM + lrow) * K + lcol;
  const float* gB1 = W1 + (size_t)(blockN + lrow) * K + lcol;
  const float* gB2 = W1 + (size_t)(4096 + blockN + lrow) * K + lcol;
  const unsigned ldsA[2] = {lds_off(&As[0][lrow * 40 + lcol]),
                            lds_off(&As[1][lrow * 40 + lcol])};

  v8f acc1[2][4], acc2[2][4];
#pragma unroll
  for (int mt = 0; mt < 2; ++mt)
#pragma unroll
    for (int nt = 0; nt < 4; ++nt)
#pragma unroll
      for (int r = 0; r < 8; ++r) { acc1[mt][nt][r] = 0.f; acc2[mt][nt][r] = 0.f; }

  async_ld16(ldsA[0], gA);
  async_ld16(ldsA[0] + 16, gA + 8);
  float4 fr1[4], fr2[4];
  {
    const float4* s1 = (const float4*)gB1;
    const float4* s2 = (const float4*)gB2;
#pragma unroll
    for (int j = 0; j < 4; ++j) { fr1[j] = s1[j]; fr2[j] = s2[j]; }
  }

  const int nk = K >> 5;
  for (int i = 0; i < nk; ++i) {
    const int cur = i & 1;
    cvt_store16(fr1, &B1s[cur][lrow * 40 + lcol]);
    cvt_store16(fr2, &B2s[cur][lrow * 40 + lcol]);
    wait_async0();
    __syncthreads();
    if (i + 1 < nk) {
      const int nxt = (i + 1) << 5;
      async_ld16(ldsA[cur ^ 1], gA + nxt);
      async_ld16(ldsA[cur ^ 1] + 16, gA + nxt + 8);
      const float4* s1 = (const float4*)(gB1 + nxt);
      const float4* s2 = (const float4*)(gB2 + nxt);
#pragma unroll
      for (int j = 0; j < 4; ++j) { fr1[j] = s1[j]; fr2[j] = s2[j]; }
    }
    Frag a0, a1;
    gather_a(As[cur], 40, waveM * 32, 0, a0);
    gather_a(As[cur], 40, waveM * 32 + 16, 0, a1);
#pragma unroll
    for (int nt = 0; nt < 4; ++nt) {
      Frag bb;
      gather_b_nk(B1s[cur], 40, waveN * 64 + nt * 16, 0, bb);
      acc1[0][nt] = wmma_bf16(a0, bb, acc1[0][nt]);
      acc1[1][nt] = wmma_bf16(a1, bb, acc1[1][nt]);
      gather_b_nk(B2s[cur], 40, waveN * 64 + nt * 16, 0, bb);
      acc2[0][nt] = wmma_bf16(a0, bb, acc2[0][nt]);
      acc2[1][nt] = wmma_bf16(a1, bb, acc2[1][nt]);
    }
    __syncthreads();
  }
#pragma unroll
  for (int mt = 0; mt < 2; ++mt)
#pragma unroll
    for (int nt = 0; nt < 4; ++nt)
#pragma unroll
      for (int r = 0; r < 8; ++r) {
        const int row = blockM + waveM * 32 + mt * 16 + r + 8 * h;
        const int col = blockN + waveN * 64 + nt * 16 + nlo;
        const float u1 = acc1[mt][nt][r];
        const float u2 = acc2[mt][nt][r];
        const float sg = u2 / (1.0f + __expf(-u2));   // silu
        G[(size_t)row * 4096 + col] = f2bf(u1 * sg);
      }
}

// ---------------------------------------------------------------- Flash attention (causal)
// Q,K,V bf16 [B,S,H*DK]; per block: one (b,head), 128-query tile; key tiles of 64.
// K/V double-buffered via async-to-LDS DMA; Q buffer is re-used for P tiles
// (safe: each wave reads its own 16 Q rows into registers before writing P there).
__global__ __launch_bounds__(256) void attn_kernel(
    const unsigned short* __restrict__ Q, const unsigned short* __restrict__ Km,
    const unsigned short* __restrict__ Vm, unsigned short* __restrict__ Ctx) {
  constexpr int S = 2048, D = 1024;
  __shared__ __align__(16) unsigned short QPs[128 * 72];   // Q staging, then P tiles
  __shared__ __align__(16) unsigned short Ks[2][64 * 72];
  __shared__ __align__(16) unsigned short Vs[2][64 * 72];

  const int t = threadIdx.x;
  const int lane = t & 31;
  const int wv = t >> 5;           // 8 waves; wave owns 16 query rows
  const int h = lane >> 4, nlo = lane & 15;
  const int wRow = wv * 16;

  const int b = blockIdx.y >> 4;
  const int head = blockIdx.y & 15;
  const int qbase = blockIdx.x * 128;

  // per-thread K/V staging coordinates: row = t>>2 (0..63), col chunk = (t&3)*16
  const int kvrow = t >> 2, kvcol = (t & 3) * 16;
  const size_t kvstat = ((size_t)(b * S)) * D + head * 64 + kvcol;
  const unsigned ldsK[2] = {lds_off(&Ks[0][kvrow * 72 + kvcol]),
                            lds_off(&Ks[1][kvrow * 72 + kvcol])};
  const unsigned ldsV[2] = {lds_off(&Vs[0][kvrow * 72 + kvcol]),
                            lds_off(&Vs[1][kvrow * 72 + kvcol])};

  // stage Q tile [128 x 64] (async) + K/V tile 0 (async)
  {
    const int row = t >> 1, c = (t & 1) * 32;
    const unsigned short* src = Q + ((size_t)(b * S + qbase + row)) * D + head * 64 + c;
    const unsigned dq = lds_off(&QPs[row * 72 + c]);
    async_ld16(dq,      src);
    async_ld16(dq + 16, src + 8);
    async_ld16(dq + 32, src + 16);
    async_ld16(dq + 48, src + 24);
  }
  {
    const unsigned short* sk = Km + kvstat + (size_t)kvrow * D;
    const unsigned short* sv = Vm + kvstat + (size_t)kvrow * D;
    async_ld16(ldsK[0], sk);  async_ld16(ldsK[0] + 16, sk + 8);
    async_ld16(ldsV[0], sv);  async_ld16(ldsV[0] + 16, sv + 8);
  }
  wait_async0();
  __syncthreads();

  Frag aq[2];
  gather_a(QPs, 72, wRow, 0, aq[0]);
  gather_a(QPs, 72, wRow, 1, aq[1]);

  float m_r[8], l_r[8];
  v8f o[4];
#pragma unroll
  for (int r = 0; r < 8; ++r) { m_r[r] = -1e30f; l_r[r] = 0.f; }
#pragma unroll
  for (int nt = 0; nt < 4; ++nt)
#pragma unroll
    for (int r = 0; r < 8; ++r) o[nt][r] = 0.f;

  const int numkt = qbase / 64 + 2;
  for (int kt = 0; kt < numkt; ++kt) {
    const int kbase = kt * 64;
    const int cur = kt & 1;
    if (kt + 1 < numkt) {  // issue next K/V tile while computing this one
      const unsigned short* sk = Km + kvstat + (size_t)((kt + 1) * 64 + kvrow) * D;
      const unsigned short* sv = Vm + kvstat + (size_t)((kt + 1) * 64 + kvrow) * D;
      async_ld16(ldsK[cur ^ 1], sk);  async_ld16(ldsK[cur ^ 1] + 16, sk + 8);
      async_ld16(ldsV[cur ^ 1], sv);  async_ld16(ldsV[cur ^ 1] + 16, sv + 8);
    }

    // scores = q @ k^T  (16 x 64 per wave)
    v8f s[4];
#pragma unroll
    for (int nt = 0; nt < 4; ++nt)
#pragma unroll
      for (int r = 0; r < 8; ++r) s[nt][r] = 0.f;
#pragma unroll
    for (int ks = 0; ks < 2; ++ks)
#pragma unroll
      for (int nt = 0; nt < 4; ++nt) {
        Frag bk;
        gather_b_nk(Ks[cur], 72, nt * 16, ks, bk);
        s[nt] = wmma_bf16(aq[ks], bk, s[nt]);
      }

    // scale + causal mask
    const bool needMask = (kbase + 63) > (qbase + wRow);
#pragma unroll
    for (int nt = 0; nt < 4; ++nt)
#pragma unroll
      for (int r = 0; r < 8; ++r) {
        float val = s[nt][r] * 0.125f;  // 1/sqrt(64)
        if (needMask) {
          const int qrow = qbase + wRow + r + 8 * h;
          const int kcol = kbase + nt * 16 + nlo;
          if (kcol > qrow) val = -1e30f;
        }
        s[nt][r] = val;
      }

    // online softmax; row stats across 16-lane groups
#pragma unroll
    for (int r = 0; r < 8; ++r) {
      float rowm = fmaxf(fmaxf(s[0][r], s[1][r]), fmaxf(s[2][r], s[3][r]));
#pragma unroll
      for (int off = 1; off < 16; off <<= 1)
        rowm = fmaxf(rowm, __shfl_xor(rowm, off, 32));
      const float mnew = fmaxf(m_r[r], rowm);
      const float alpha = __expf(m_r[r] - mnew);
      m_r[r] = mnew;
      float rsum = 0.f;
#pragma unroll
      for (int nt = 0; nt < 4; ++nt) {
        const float p = __expf(s[nt][r] - mnew);
        s[nt][r] = p;
        rsum += p;
      }
#pragma unroll
      for (int off = 1; off < 16; off <<= 1)
        rsum += __shfl_xor(rsum, off, 32);
      l_r[r] = l_r[r] * alpha + rsum;
#pragma unroll
      for (int nt = 0; nt < 4; ++nt) o[nt][r] *= alpha;
    }

    // p -> LDS (bf16) to re-fragment as A matrix (wave-private rows of QPs)
#pragma unroll
    for (int nt = 0; nt < 4; ++nt)
#pragma unroll
      for (int r = 0; r < 8; ++r)
        QPs[(wRow + r + 8 * h) * 72 + nt * 16 + nlo] = f2bf(s[nt][r]);
    __syncthreads();

    // o += p @ v
#pragma unroll
    for (int ks = 0; ks < 2; ++ks) {
      Frag ap;
      gather_a(QPs, 72, wRow, ks, ap);
#pragma unroll
      for (int nt = 0; nt < 4; ++nt) {
        Frag bv;
        gather_b_kn(Vs[cur], 72, nt * 16, ks, bv);
        o[nt] = wmma_bf16(ap, bv, o[nt]);
      }
    }
    if (kt + 1 < numkt) wait_async0();  // next tile landed
    __syncthreads();
  }

  // normalize + store ctx (bf16)
#pragma unroll
  for (int nt = 0; nt < 4; ++nt)
#pragma unroll
    for (int r = 0; r < 8; ++r) {
      const int row = qbase + wRow + r + 8 * h;
      const float val = o[nt][r] / l_r[r];
      Ctx[((size_t)(b * S + row)) * D + head * 64 + nt * 16 + nlo] = f2bf(val);
    }
}

// ---------------------------------------------------------------- launch
extern "C" void kernel_launch(void* const* d_in, const int* in_sizes, int n_in,
                              void* d_out, int out_size, void* d_ws, size_t ws_size,
                              hipStream_t stream) {
  (void)in_sizes; (void)n_in; (void)out_size; (void)ws_size;
  const float* x  = (const float*)d_in[0];
  // d_in[1] = causal mask (unused; causality handled analytically)
  const float* wq = (const float*)d_in[2];
  const float* wk = (const float*)d_in[3];
  const float* wv = (const float*)d_in[4];
  const float* wo = (const float*)d_in[5];
  const float* w1 = (const float*)d_in[6];
  const float* w2 = (const float*)d_in[7];
  const float* g1 = (const float*)d_in[8];
  const float* g2 = (const float*)d_in[9];

  constexpr int Bz = 2, S = 2048, D = 1024, DFF = 4096;
  const int M = Bz * S;  // 4096

  char* p = (char*)d_ws;
  auto alloc = [&](size_t bytes) {
    char* r = p;
    p += (bytes + 255) & ~(size_t)255;
    return r;
  };
  unsigned short* hb   = (unsigned short*)alloc((size_t)M * D * 2);    // rmsnorm1(x)
  unsigned short* qb   = (unsigned short*)alloc((size_t)M * D * 2);    // q (post-rope)
  unsigned short* kb   = (unsigned short*)alloc((size_t)M * D * 2);    // k (post-rope)
  unsigned short* vb   = (unsigned short*)alloc((size_t)M * D * 2);    // v
  unsigned short* ctxb = (unsigned short*)alloc((size_t)M * D * 2);    // attention output
  float*          x1   = (float*)alloc((size_t)M * D * 4);             // x + ctx@wo^T
  unsigned short* h2b  = (unsigned short*)alloc((size_t)M * D * 2);    // rmsnorm2(x1)
  unsigned short* gb   = (unsigned short*)alloc((size_t)M * DFF * 2);  // swiglu output

  const dim3 gD(D / 128, M / 128);     // (8, 32)
  const dim3 gF(DFF / 128, M / 128);   // (32, 32)
  const dim3 gA(S / 128, Bz * 16);     // (16, 32)

  // 1. h = rmsnorm(x, g1)
  rmsnorm_kernel<<<M, 256, 0, stream>>>(x, g1, hb);
  // 2. q,k,v projections (bf16 out)
  gemm_bf16_kernel<0><<<gD, 256, 0, stream>>>(hb, wq, qb, nullptr, M, D, D);
  gemm_bf16_kernel<0><<<gD, 256, 0, stream>>>(hb, wk, kb, nullptr, M, D, D);
  gemm_bf16_kernel<0><<<gD, 256, 0, stream>>>(hb, wv, vb, nullptr, M, D, D);
  // 3. RoPE in-place on q,k
  rope_kernel<<<(Bz * S * 16 * 32) / 256, 256, 0, stream>>>(qb, kb);
  // 4. causal flash attention -> ctx
  attn_kernel<<<gA, 256, 0, stream>>>(qb, kb, vb, ctxb);
  // 5. x1 = x + ctx @ wo^T
  gemm_bf16_kernel<1><<<gD, 256, 0, stream>>>(ctxb, wo, x1, x, M, D, D);
  // 6. h2 = rmsnorm(x1, g2)
  rmsnorm_kernel<<<M, 256, 0, stream>>>(x1, g2, h2b);
  // 7. g = u1 * silu(u2), u = h2 @ w1^T (fused)
  gemm_swiglu_kernel<<<gF, 256, 0, stream>>>(h2b, w1, gb, M, D);
  // 8. out = x1 + g @ w2^T
  gemm_bf16_kernel<1><<<gD, 256, 0, stream>>>(gb, w2, d_out, x1, M, D, DFF);
}
